// NeuralSFA_29377576305410
// MI455X (gfx1250) — matrix operand order repaired and checked by reading, not compiled, not run on hardware
//
#include <hip/hip_runtime.h>

typedef __attribute__((ext_vector_type(16))) _Float16 v16h;
typedef __attribute__((ext_vector_type(8)))  _Float16 v8h;
typedef __attribute__((ext_vector_type(8)))  float    v8f;

#define TPB 256

// ---------------------------------------------------------------------------
// Fused per-image CNN. Data path:
//   image --(global_load_async_to_lds_b128)--> LDS
//   conv1+relu+pool (fp32 VALU, K=9 too thin for WMMA)  -> s_x1 (NHWC f16)
//   im2col staged in WMMA fragment layout (v8h LDS copies)
//   conv2: v_wmma_f32_16x16x32_f16, M=16 oc x N=121 pos (8 waves) x K=72->96
//   conv3: v_wmma_f32_16x16x32_f16, M=32 oc (2 waves) x N=9 pos x K=144->160
//   dense 32->10 + softmax (fp32)
// Only 3136 B in / 80 B out per image touch HBM (roofline ~1.1us total).
// ---------------------------------------------------------------------------
__global__ __launch_bounds__(TPB)
void sfa_cnn_kernel(const float* __restrict__ seq,
                    const float* __restrict__ w1, const float* __restrict__ b1,
                    const float* __restrict__ w2, const float* __restrict__ b2,
                    const float* __restrict__ w3, const float* __restrict__ b3,
                    const float* __restrict__ dw, const float* __restrict__ db,
                    float* __restrict__ digit_out, float* __restrict__ guard_out)
{
    __shared__ alignas(16) float    s_img[28*28];
    __shared__ alignas(32) _Float16 s_x1[13*13*8];      // pooled conv1, NHWC
    __shared__ float    s_w1[8*9];
    __shared__ float    s_b1[8];
    __shared__ alignas(32) _Float16 s_Af2[3*512];       // conv2 A frags [ks][lane][16]
    __shared__ alignas(32) _Float16 s_Bf2[8*3*512];     // conv2 B frags [ntile][ks][lane][16]
    __shared__ float    s_b2[16];
    __shared__ float    s_c2[16*128];                   // conv2 relu'd [oc][pos pad128]
    __shared__ alignas(32) _Float16 s_x2[5*5*16];       // pooled conv2, NHWC
    __shared__ alignas(32) _Float16 s_Af3[2*5*512];     // conv3 A frags [mtile][ks][lane][16]
    __shared__ alignas(32) _Float16 s_Bf3[5*512];       // conv3 B frags [ks][lane][16]
    __shared__ float    s_b3[32];
    __shared__ float    s_c3[32*16];                    // conv3 relu'd [oc][pos pad16]
    __shared__ float    s_x3[32];
    __shared__ float    s_dw[10*32];
    __shared__ float    s_db[10];
    __shared__ float    s_log[10];
    __shared__ float    s_exp[10];

    const int tid  = threadIdx.x;
    const int lane = tid & 31;
    const int wave = tid >> 5;
    const long t   = blockIdx.x;

    // ---- stage 0a: async DMA of the image into LDS (196 x 16B) -------------
    if (tid < 196) {
        unsigned lds_off = (unsigned)(unsigned long long)&s_img[tid * 4];
        unsigned long long ga =
            (unsigned long long)(seq + t * 784 + tid * 4);
        asm volatile("global_load_async_to_lds_b128 %0, %1, off"
                     :: "v"(lds_off), "v"(ga) : "memory");
    }

    // ---- stage 0b: weights, pre-scattered into WMMA A-fragment layout ------
    for (int i = tid; i < 72; i += TPB) s_w1[i] = w1[i];
    if (tid < 8)  s_b1[tid] = b1[tid];
    if (tid < 16) s_b2[tid] = b2[tid];
    if (tid < 32) s_b3[tid] = b3[tid];
    for (int i = tid; i < 320; i += TPB) s_dw[i] = dw[i];
    if (tid < 10) s_db[tid] = db[tid];

    // conv2 A frags: oc = lane&15; e->K per ISA A layout; K = (dy*3+dx)*8+ic
    for (int i = tid; i < 3 * 512; i += TPB) {
        int ks = i >> 9, r = i & 511, ln = r >> 4, e = r & 15;
        int oc = ln & 15;
        int krel = (ln < 16) ? ((e < 8) ? e : e + 8) : ((e < 8) ? e + 8 : e + 16);
        int K = ks * 32 + krel;
        _Float16 v = (_Float16)0.0f;
        if (K < 72) { int dydx = K >> 3, ic = K & 7; v = (_Float16)w2[oc*72 + ic*9 + dydx]; }
        s_Af2[i] = v;
    }
    // conv3 A frags: oc = mtile*16 + (lane&15); K = (dy*3+dx)*16+ic
    for (int i = tid; i < 2 * 5 * 512; i += TPB) {
        int blk = i >> 9, r = i & 511, ln = r >> 4, e = r & 15;
        int mt = blk / 5;
        int oc = mt * 16 + (ln & 15);
        int krel = (ln < 16) ? ((e < 8) ? e : e + 8) : ((e < 8) ? e + 8 : e + 16);
        int K = (blk % 5) * 32 + krel;
        _Float16 v = (_Float16)0.0f;
        if (K < 144) { int dydx = K >> 4, ic = K & 15; v = (_Float16)w3[oc*144 + ic*9 + dydx]; }
        s_Af3[i] = v;
    }
    asm volatile("s_wait_asynccnt 0" ::: "memory");
    __syncthreads();

    // ---- stage 1: conv1 + relu + 2x2 avgpool -> s_x1 NHWC f16 --------------
    for (int idx = tid; idx < 13 * 13 * 8; idx += TPB) {
        int oc = idx & 7, p = idx >> 3, px = p % 13, py = p / 13;
        float acc = 0.f;
        #pragma unroll
        for (int r = 0; r < 2; ++r)
        #pragma unroll
        for (int c = 0; c < 2; ++c) {
            int y = 2 * py + r, x = 2 * px + c;
            float s = s_b1[oc];
            #pragma unroll
            for (int dy = 0; dy < 3; ++dy)
            #pragma unroll
            for (int dx = 0; dx < 3; ++dx)
                s += s_img[(y + dy) * 28 + (x + dx)] * s_w1[oc * 9 + dy * 3 + dx];
            acc += fmaxf(s, 0.f);
        }
        s_x1[idx] = (_Float16)(acc * 0.25f);
    }
    __syncthreads();

    // ---- stage 2a: im2col B for conv2 in fragment layout (v8h copies) ------
    // Aligned groups of 8 K share one (dy,dx) -> 8 contiguous NHWC channels.
    for (int i = tid; i < (8 * 3 * 512) / 8; i += TPB) {
        int idx8 = i * 8;
        int blk = idx8 >> 9;                      // ntile*3 + ks
        int w = blk / 3, ks = blk % 3;
        int r = idx8 & 511, ln = r >> 4, e0 = r & 15;   // e0 in {0,8}
        int n  = w * 16 + (ln & 15);
        int K0 = ks * 32 + ((ln < 16) ? 2 * e0 : 8 + 2 * e0);  // multiple of 8
        v8h v = {};
        if (n < 121 && K0 < 72) {
            int dydx = K0 >> 3, dy = dydx / 3, dx = dydx % 3;
            int oy = n / 11, ox = n % 11;
            v = *(const v8h*)&s_x1[((oy + dy) * 13 + (ox + dx)) * 8];
        }
        *(v8h*)&s_Bf2[idx8] = v;
    }
    __syncthreads();

    // ---- stage 2b: conv2 WMMA, one 16-pos tile per wave --------------------
    {
        const int n0 = wave * 16;
        const int n  = n0 + (lane & 15);
        v8f c = {};
        #pragma unroll
        for (int ks = 0; ks < 3; ++ks) {
            v16h a = ((const v16h*)(s_Af2 + ks * 512))[lane];
            v16h b = ((const v16h*)(s_Bf2 + (wave * 3 + ks) * 512))[lane];
            c = __builtin_amdgcn_wmma_f32_16x16x32_f16(
                    false, a, false, b, (short)0, c, false, false);
        }
        #pragma unroll
        for (int r = 0; r < 8; ++r) {             // D: VGPR r -> M = r + 8*(lane>=16)
            int M = r + ((lane < 16) ? 0 : 8);
            s_c2[M * 128 + n] = fmaxf(c[r] + s_b2[M], 0.f);
        }
    }
    __syncthreads();

    // pool 11x11 -> 5x5, pack NHWC f16
    for (int idx = tid; idx < 5 * 5 * 16; idx += TPB) {
        int oc = idx & 15, p = idx >> 4, px = p % 5, py = p / 5;
        const float* row = &s_c2[oc * 128];
        float v = 0.25f * (row[(2*py)*11 + 2*px]     + row[(2*py)*11 + 2*px + 1] +
                           row[(2*py+1)*11 + 2*px]   + row[(2*py+1)*11 + 2*px + 1]);
        s_x2[idx] = (_Float16)v;
    }
    __syncthreads();

    // ---- stage 3a: im2col B for conv3 (shared by both M-tiles) -------------
    for (int i = tid; i < (5 * 512) / 8; i += TPB) {
        int idx8 = i * 8;
        int ks = idx8 >> 9;
        int r = idx8 & 511, ln = r >> 4, e0 = r & 15;
        int n  = ln & 15;
        int K0 = ks * 32 + ((ln < 16) ? 2 * e0 : 8 + 2 * e0);
        v8h v = {};
        if (n < 9 && K0 < 144) {
            int dydx = K0 >> 4, dy = dydx / 3, dx = dydx % 3;
            int oy = n / 3, ox = n % 3;
            v = *(const v8h*)&s_x2[((oy + dy) * 5 + (ox + dx)) * 16 + (K0 & 15)];
        }
        *(v8h*)&s_Bf3[idx8] = v;
    }
    __syncthreads();

    // ---- stage 3b: conv3 WMMA, waves 0,1 cover oc 0-15 / 16-31 -------------
    if (wave < 2) {
        const int n = lane & 15;
        v8f c = {};
        #pragma unroll
        for (int ks = 0; ks < 5; ++ks) {
            v16h a = ((const v16h*)(s_Af3 + (wave * 5 + ks) * 512))[lane];
            v16h b = ((const v16h*)(s_Bf3 + ks * 512))[lane];
            c = __builtin_amdgcn_wmma_f32_16x16x32_f16(
                    false, a, false, b, (short)0, c, false, false);
        }
        #pragma unroll
        for (int r = 0; r < 8; ++r) {
            int M = wave * 16 + r + ((lane < 16) ? 0 : 8);
            s_c3[M * 16 + n] = fmaxf(c[r] + s_b3[M], 0.f);
        }
    }
    __syncthreads();

    // 3x3 -> 1x1 avgpool uses conv positions n = 0,1,3,4
    if (tid < 32) {
        const float* row = &s_c3[tid * 16];
        s_x3[tid] = 0.25f * (row[0] + row[1] + row[3] + row[4]);
    }
    __syncthreads();

    // ---- stage 4: dense 32->10 + softmax -----------------------------------
    if (tid < 10) {
        float acc = s_db[tid];
        #pragma unroll
        for (int c2 = 0; c2 < 32; ++c2) acc += s_x3[c2] * s_dw[tid * 32 + c2];
        s_log[tid] = acc;
    }
    __syncthreads();
    if (tid < 10) {
        float m = s_log[0];
        for (int j = 1; j < 10; ++j) m = fmaxf(m, s_log[j]);
        s_exp[tid] = expf(s_log[tid] - m);
    }
    __syncthreads();
    if (tid < 10) {
        float s = 0.f;
        for (int j = 0; j < 10; ++j) s += s_exp[j];
        float p = s_exp[tid] / s;
        digit_out[t * 10 + tid] = p;
        guard_out[t * 10 + tid] = p;
    }
}

// ---------------------------------------------------------------------------
// complete[g][i][j]: softmax(trans/0.1) rows + accepting row e9
// ---------------------------------------------------------------------------
__global__ __launch_bounds__(128)
void sfa_trans_kernel(const float* __restrict__ trans, float* __restrict__ complete)
{
    int tid = threadIdx.x;
    if (tid < 90) {
        int g = tid / 9, i = tid % 9;
        const float* row = trans + (g * 9 + i) * 10;
        float v[10];
        float m = -1e30f;
        for (int j = 0; j < 10; ++j) { v[j] = row[j] * 10.f; m = fmaxf(m, v[j]); }
        float s = 0.f;
        for (int j = 0; j < 10; ++j) { v[j] = expf(v[j] - m); s += v[j]; }
        for (int j = 0; j < 10; ++j) complete[g * 100 + i * 10 + j] = v[j] / s;
    } else if (tid < 100) {
        int g = tid - 90;
        for (int j = 0; j < 10; ++j) complete[g * 100 + 90 + j] = (j == 9) ? 1.f : 0.f;
    }
}

// ---------------------------------------------------------------------------
// Per-chunk ordered product of 10x10 effective matrices (associative chain)
// ---------------------------------------------------------------------------
__global__ __launch_bounds__(128)
void sfa_chunk_kernel(const float* __restrict__ gp, const float* __restrict__ complete,
                      float* __restrict__ part, int T, int chunk)
{
    __shared__ float s_P[100], s_E[100], s_gp[10];
    const int tid = threadIdx.x;
    const long b = blockIdx.x;
    if (tid < 100) s_P[tid] = (tid / 10 == tid % 10) ? 1.f : 0.f;
    __syncthreads();
    for (int s = 0; s < chunk; ++s) {
        long t = b * (long)chunk + s;
        if (tid < 10) s_gp[tid] = (t < T) ? gp[t * 10 + tid] : 0.f;
        __syncthreads();
        if (tid < 100) {
            int k = tid / 10, j = tid % 10;
            float e;
            if (t < T) {
                e = 0.f;
                for (int g = 0; g < 10; ++g) e += s_gp[g] * complete[g * 100 + k * 10 + j];
            } else {
                e = (k == j) ? 1.f : 0.f;
            }
            s_E[tid] = e;
        }
        __syncthreads();
        float np = 0.f;
        if (tid < 100) {
            int i = tid / 10, j = tid % 10;
            for (int k = 0; k < 10; ++k) np += s_P[i * 10 + k] * s_E[k * 10 + j];
        }
        __syncthreads();
        if (tid < 100) s_P[tid] = np;
        __syncthreads();
    }
    if (tid < 100) part[b * 100 + tid] = s_P[tid];
}

// ---------------------------------------------------------------------------
// Ordered fold of chunk partials: final = e0 * P_0 * P_1 * ...
// ---------------------------------------------------------------------------
__global__ __launch_bounds__(32)
void sfa_fold_kernel(const float* __restrict__ part, float* __restrict__ final_out,
                     int nchunk)
{
    __shared__ float s_v[10];
    const int tid = threadIdx.x;
    if (tid < 10) s_v[tid] = (tid == 0) ? 1.f : 0.f;
    __syncthreads();
    for (int b = 0; b < nchunk; ++b) {
        float nv = 0.f;
        if (tid < 10)
            for (int k = 0; k < 10; ++k) nv += s_v[k] * part[b * 100 + k * 10 + tid];
        __syncthreads();
        if (tid < 10) s_v[tid] = nv;
        __syncthreads();
    }
    if (tid < 10) final_out[tid] = s_v[tid];
}

extern "C" void kernel_launch(void* const* d_in, const int* in_sizes, int n_in,
                              void* d_out, int out_size, void* d_ws, size_t ws_size,
                              hipStream_t stream)
{
    const float* seq = (const float*)d_in[0];
    const float* w1  = (const float*)d_in[1];
    const float* b1  = (const float*)d_in[2];
    const float* w2  = (const float*)d_in[3];
    const float* b2  = (const float*)d_in[4];
    const float* w3  = (const float*)d_in[5];
    const float* b3  = (const float*)d_in[6];
    const float* dw  = (const float*)d_in[7];
    const float* db  = (const float*)d_in[8];
    const float* tr  = (const float*)d_in[9];

    const int T = in_sizes[0] / (28 * 28);

    float* out    = (float*)d_out;
    float* digit  = out;
    float* guard  = out + (long)T * 10;
    float* finals = out + (long)2 * T * 10;

    float* complete = (float*)d_ws;          // 1000 floats
    float* part     = complete + 1024;       // nchunk*100 floats

    const int CH = 64;
    const int nchunk = (T + CH - 1) / CH;

    sfa_cnn_kernel<<<T, TPB, 0, stream>>>(seq, w1, b1, w2, b2, w3, b3, dw, db,
                                          digit, guard);
    sfa_trans_kernel<<<1, 128, 0, stream>>>(tr, complete);
    sfa_chunk_kernel<<<nchunk, 128, 0, stream>>>(digit, complete, part, T, CH);
    sfa_fold_kernel<<<1, 32, 0, stream>>>(part, finals, nchunk);
}
